// BoundaryLoss_9569187135492
// MI455X (gfx1250) — matrix operand order repaired and checked by reading, not compiled, not run on hardware
//
#include <hip/hip_runtime.h>

// CDNA5 (gfx1250) wave32 WMMA types
typedef __attribute__((ext_vector_type(16))) _Float16 v16h;
typedef __attribute__((ext_vector_type(8)))  _Float16 v8h;
typedef __attribute__((ext_vector_type(8)))  float    v8f;

#define DVOL    128
#define S_PLANE (DVOL * DVOL)          // 16384
#define S_VOL   (DVOL * DVOL * DVOL)   // 2097152
#define NVOX    (2 * S_VOL)            // 4194304 voxels (B=2)
#define NCLS    11
#define TILE    16
#define CHUNK   32                     // d-slices per block
#define NBLK_B  (2 * 8 * 8 * 4)        // batch * tilesH * tilesW * dChunks = 512
#define CSTRIDE 40                     // padded column stride (halves): 80 B = 20 banks

// Concatenate two contiguous 16B LDS loads into one WMMA operand register set.
__device__ __forceinline__ v16h load_col16(const _Float16* p) {
    v8h lo = *(const v8h*)(p);
    v8h hi = *(const v8h*)(p + 8);
    return __builtin_shufflevector(lo, hi, 0, 1, 2, 3, 4, 5, 6, 7,
                                           8, 9, 10, 11, 12, 13, 14, 15);
}

// ---------------------------------------------------------------------------
// Pass 1: streaming argmax over 11 classes + compaction of pred/target to u8.
// Moves the 184.5 MB of logits once (HBM-bound, ~9 us floor at 23.3 TB/s) and
// leaves two 4.2 MB u8 fields that stay resident in the 192 MB L2.
// float4-vectorized: 4 voxels/thread, 11 coalesced 16B loads each.
// ---------------------------------------------------------------------------
__global__ __launch_bounds__(256)
void argmax_pack_kernel(const float* __restrict__ logits,
                        const int*   __restrict__ target,
                        unsigned char* __restrict__ pred8,
                        unsigned char* __restrict__ targ8)
{
    size_t t  = (size_t)blockIdx.x * blockDim.x + threadIdx.x;
    size_t v0 = t * 4;
    if (v0 >= (size_t)NVOX) return;
    size_t b = v0 >> 21;               // S_VOL = 2^21
    size_t s = v0 & (size_t)(S_VOL - 1);

    const float4* p0 = reinterpret_cast<const float4*>(
        logits + (b * NCLS) * (size_t)S_VOL + s);
    const size_t stride4 = (size_t)S_VOL / 4;   // class stride in float4 units

    float4 bv = p0[0];
    int ix = 0, iy = 0, iz = 0, iw = 0;
    #pragma unroll
    for (int c = 1; c < NCLS; ++c) {
        float4 v = p0[(size_t)c * stride4];
        if (v.x > bv.x) { bv.x = v.x; ix = c; }   // strict > keeps first max
        if (v.y > bv.y) { bv.y = v.y; iy = c; }   // (matches jnp.argmax)
        if (v.z > bv.z) { bv.z = v.z; iz = c; }
        if (v.w > bv.w) { bv.w = v.w; iw = c; }
    }
    *reinterpret_cast<uchar4*>(pred8 + v0) =
        make_uchar4((unsigned char)ix, (unsigned char)iy,
                    (unsigned char)iz, (unsigned char)iw);

    int4 tg = *reinterpret_cast<const int4*>(target + v0);
    *reinterpret_cast<uchar4*>(targ8 + v0) =
        make_uchar4((unsigned char)tg.x, (unsigned char)tg.y,
                    (unsigned char)tg.z, (unsigned char)tg.w);
}

// ---------------------------------------------------------------------------
// Pass 2: fused boundary stencil + weighted squared diff.
// Stencil factorization: conv(x) = (32*x - S_d S_h S_w x) / 24, S = [1,2,1].
//   - S_w: packed v_pk f16 adds over column-major LDS tile (exact: <= 40)
//   - S_h: one v_wmma_f32_16x16x32_f16 per field per slice, A = constant
//          16x32 band matrix (1,2,1 on rows 0..17; K 18..31 zero-padded via
//          the zero-initialized LDS column tail)
//   - S_d: rolling 3-slice window, elementwise on the WMMA f32 D layout
//          (lane L, vgpr v -> h = v + 8*(L>=16), w = L&15) -- no relayout.
// Column-major LDS (stride 40 halves = 20 banks) makes each lane's B operand
// two contiguous conflict-free ds_load_b128 per tap column, in the contiguous (A-matching)
// B element->K mapping (element i <-> K = i + 16*half) with no repacking.
// One wave32 per 16x16 (H,W) tile; EXEC all-ones at WMMA as required.
// ---------------------------------------------------------------------------
__global__ __launch_bounds__(32)
void boundary_loss_kernel(const unsigned char* __restrict__ pred8,
                          const unsigned char* __restrict__ targ8,
                          float* __restrict__ partials)
{
    // X[col][row]: cols 0..17 (w halo), rows 0..17 (h halo), rows 18..39 = 0
    __shared__ __attribute__((aligned(16))) _Float16 Xp[18][CSTRIDE];
    __shared__ __attribute__((aligned(16))) _Float16 Xt[18][CSTRIDE];

    const int lane = threadIdx.x;     // 0..31, one wave
    const int half = lane >> 4;       // lane group (ISA WMMA layout)
    const int n    = lane & 15;       // B column / A row / output w within tile
    const int kofs = 16 * half;       // this lane's K-row window base

    // block decomposition: 8x8 HW tiles, 4 d-chunks, 2 batches
    const int bi  = blockIdx.x;
    const int w0  = (bi & 7) * TILE;
    const int h0  = ((bi >> 3) & 7) * TILE;
    const int dLo = ((bi >> 6) & 3) * CHUNK;
    const int b   = bi >> 8;
    const size_t bBase = (size_t)b * S_VOL;

    // Zero the whole LDS tile once: gives the K >= 18 zero padding for free.
    {
        float4  z4 = make_float4(0.f, 0.f, 0.f, 0.f);
        float4* zp = (float4*)&Xp[0][0];
        float4* zt = (float4*)&Xt[0][0];
        const int n16 = (18 * CSTRIDE * 2) / 16;   // 90 float4 per field
        for (int i = lane; i < n16; i += 32) { zp[i] = z4; zt[i] = z4; }
    }

    // Constant band A (16x32 f16). A layout matching B: lane half 0 holds
    // K = {0..15}, half 1 holds K = {16..31}; element i -> K:
    //   K = i + 16*half
    v16h Amat;
    #pragma unroll
    for (int i = 0; i < 16; ++i) {
        int K = i + 16 * half;
        float val = (K == n) ? 1.f : (K == n + 1) ? 2.f : (K == n + 2) ? 1.f : 0.f;
        Amat[i] = (_Float16)val;
    }

    v8f Um2p = {}, Um1p = {}, Um2t = {}, Um1t = {};
    float cm1p[8], cm1t[8];
    #pragma unroll
    for (int v = 0; v < 8; ++v) { cm1p[v] = 0.f; cm1t[v] = 0.f; }
    float acc = 0.f;

    for (int dd = dLo - 1; dd <= dLo + CHUNK; ++dd) {
        __syncthreads();
        // stage raw slice dd (zero-padded halo) into LDS, column-major
        const bool dok = (dd >= 0) && (dd < DVOL);
        for (int idx = lane; idx < 18 * 18; idx += 32) {
            int r = idx / 18, c = idx - r * 18;
            int h = h0 - 1 + r, w = w0 - 1 + c;
            unsigned char pv = 0, tv = 0;
            if (dok && h >= 0 && h < DVOL && w >= 0 && w < DVOL) {
                size_t gi = bBase + (size_t)dd * S_PLANE + (size_t)h * DVOL + w;
                pv = pred8[gi];
                tv = targ8[gi];
            }
            Xp[c][r] = (_Float16)(float)pv;
            Xt[c][r] = (_Float16)(float)tv;
        }
        __syncthreads();

        // B operands (32x16 f16): W-direction [1,2,1] pass, fully packed.
        // load_col16 element i = row (kofs + i) of the tap column, which is
        // the contiguous K = i + 16*half order for column n, as in A.
        v16h p0 = load_col16(&Xp[n    ][kofs]);
        v16h p1 = load_col16(&Xp[n + 1][kofs]);
        v16h p2 = load_col16(&Xp[n + 2][kofs]);
        v16h t0 = load_col16(&Xt[n    ][kofs]);
        v16h t1 = load_col16(&Xt[n + 1][kofs]);
        v16h t2 = load_col16(&Xt[n + 2][kofs]);
        v16h Bp = p0 + p1 + p1 + p2;   // v_pk_add_f16, exact (<= 40)
        v16h Bt = t0 + t1 + t1 + t2;

        // centers of this slice in the D-matrix register layout
        float ccp[8], cct[8];
        #pragma unroll
        for (int v = 0; v < 8; ++v) {
            int r = v + 8 * half + 1;
            ccp[v] = (float)Xp[n + 1][r];
            cct[v] = (float)Xt[n + 1][r];
        }

        // H-direction pass: U = band(16x32) x T(32x16), f32 accumulate
        v8f zero = {};
        v8f Up = __builtin_amdgcn_wmma_f32_16x16x32_f16(
            false, Amat, false, Bp, (short)0, zero, false, false);
        v8f Ut = __builtin_amdgcn_wmma_f32_16x16x32_f16(
            false, Amat, false, Bt, (short)0, zero, false, false);

        // D-direction pass + loss for output slice d = dd-1
        if (dd >= dLo + 1) {
            #pragma unroll
            for (int v = 0; v < 8; ++v) {
                float sp = Um2p[v] + 2.f * Um1p[v] + Up[v];
                float st = Um2t[v] + 2.f * Um1t[v] + Ut[v];
                float convp = (32.f * cm1p[v] - sp) * (1.f / 24.f);
                float convt = (32.f * cm1t[v] - st) * (1.f / 24.f);
                float pb = fabsf(convp);
                float tb = fabsf(convt);
                int   cls = (int)(cm1t[v] + 0.5f);
                // SMALL_CLASSES = {2,3,5,6,7,8,9,10}; complement is {0,1,4}
                float wgt = (cls == 0 || cls == 1 || cls == 4) ? 1.f : 3.f;
                float dq = (pb - tb) * wgt;
                acc += dq * dq;
            }
        }
        Um2p = Um1p; Um1p = Up;
        Um2t = Um1t; Um1t = Ut;
        #pragma unroll
        for (int v = 0; v < 8; ++v) { cm1p[v] = ccp[v]; cm1t[v] = cct[v]; }
    }

    // wave32 reduction, deterministic per block
    #pragma unroll
    for (int off = 16; off > 0; off >>= 1)
        acc += __shfl_xor(acc, off, 32);
    if (lane == 0) partials[blockIdx.x] = acc;
}

// ---------------------------------------------------------------------------
// Pass 3: deterministic final reduction of 512 partials -> mean (no atomics).
// ---------------------------------------------------------------------------
__global__ __launch_bounds__(32)
void final_reduce_kernel(const float* __restrict__ partials,
                         float* __restrict__ out)
{
    const int lane = threadIdx.x;
    float acc = 0.f;
    for (int i = lane; i < NBLK_B; i += 32) acc += partials[i];
    #pragma unroll
    for (int off = 16; off > 0; off >>= 1)
        acc += __shfl_xor(acc, off, 32);
    if (lane == 0) out[0] = acc * (1.f / (float)NVOX);
}

extern "C" void kernel_launch(void* const* d_in, const int* in_sizes, int n_in,
                              void* d_out, int out_size, void* d_ws, size_t ws_size,
                              hipStream_t stream)
{
    const float* logits = (const float*)d_in[0];   // [2,11,128,128,128] f32
    const int*   target = (const int*)d_in[1];     // [2,1,128,128,128] i32

    unsigned char* pred8    = (unsigned char*)d_ws;          // 4 MiB
    unsigned char* targ8    = pred8 + (size_t)NVOX;          // 4 MiB
    float*         partials = (float*)(targ8 + (size_t)NVOX);// 2 KiB

    // Pass 1: 4.19M voxels / 4 per thread / 256 per block = 4096 blocks
    argmax_pack_kernel<<<NVOX / 4 / 256, 256, 0, stream>>>(
        logits, target, pred8, targ8);

    // Pass 2: one wave32 per 16x16 HW tile per d-chunk
    boundary_loss_kernel<<<NBLK_B, 32, 0, stream>>>(pred8, targ8, partials);

    // Pass 3: scalar mean
    final_reduce_kernel<<<1, 32, 0, stream>>>(partials, (float*)d_out);
}
